// BaseLSTM_4372276707596
// MI455X (gfx1250) — compile-verified, matrix operand order checked by reading, not run on hardware
//
#include <hip/hip_runtime.h>
#include <hip/hip_bf16.h>

// Problem dims
#define B_    64
#define INCH  16
#define H_    256
#define FDIM_ 128
#define NF_   1001
#define NFP_  1024     // sequence padded to multiple of 16 for WMMA tiles
#define G4H_  1024     // 4*H
#define LIN0_ 144      // FDIM + IN_CH

typedef _Float16 h8  __attribute__((ext_vector_type(8)));
typedef _Float16 h16 __attribute__((ext_vector_type(16)));
typedef float    f8  __attribute__((ext_vector_type(8)));

// ---------------------------------------------------------------------------
// Kernel 1: convert f (zero-padded to 1024 rows) and W_ih0[:, 0:128] to f16
// ---------------------------------------------------------------------------
__global__ void cvt_f16_kernel(const float* __restrict__ f,
                               const float* __restrict__ Wih0,
                               _Float16* __restrict__ f16f,
                               _Float16* __restrict__ f16w) {
    int idx = blockIdx.x * blockDim.x + threadIdx.x;   // 0 .. 2*131072-1
    if (idx >= 2 * NFP_ * FDIM_) return;
    int half = idx >> 17;            // which matrix (131072 = 2^17 elems each)
    int e    = idx & 131071;
    int row  = e >> 7;               // 128 cols
    int col  = e & 127;
    if (half == 0) {
        float v = (row < NF_) ? f[row * FDIM_ + col] : 0.0f;
        f16f[e] = (_Float16)v;
    } else {
        f16w[e] = (_Float16)Wih0[row * LIN0_ + col];   // first 128 cols of W_ih0
    }
}

// ---------------------------------------------------------------------------
// Kernel 2: Gf[t,g] = sum_k f16f[t,k] * f16w[g,k]   (M=N=1024, K=128)
// One 16x16 tile per wave32, K loop of 4 x WMMA f32_16x16x32_f16.
// A-layout per ISA: lane m = lane%16, khalf = lane/16; lane holds two
// contiguous runs of 8 halves: [khalf*8, +8) and [16+khalf*8, +8).
// ---------------------------------------------------------------------------
__global__ void __launch_bounds__(256)
wmma_gemm_kernel(const _Float16* __restrict__ A,     // [1024,128] row-major (t,k)
                 const _Float16* __restrict__ Bw,    // [1024,128] row-major (g,k)
                 float* __restrict__ C) {            // [1024,1024] (t,g)
    int wave = (blockIdx.x * blockDim.x + threadIdx.x) >> 5;  // 0..4095
    int lane = threadIdx.x & 31;
    int mt = wave >> 6;                 // 0..63 tile row (t)
    int nt = wave & 63;                 // 0..63 tile col (g)
    int mrow  = (mt << 4) + (lane & 15);
    int nrow  = (nt << 4) + (lane & 15);
    int khalf = (lane >> 4) << 3;       // 0 or 8

    f8 acc = {};
#pragma unroll
    for (int kk = 0; kk < 4; ++kk) {
        int kb = kk << 5;               // K base
        const _Float16* pA = A  + mrow * FDIM_ + kb + khalf;
        const _Float16* pB = Bw + nrow * FDIM_ + kb + khalf;
        h8 a0 = *(const h8*)(pA);
        h8 a1 = *(const h8*)(pA + 16);
        h8 b0 = *(const h8*)(pB);
        h8 b1 = *(const h8*)(pB + 16);
        h16 av, bv;
#pragma unroll
        for (int i = 0; i < 8; ++i) {
            av[i] = a0[i]; av[i + 8] = a1[i];
            bv[i] = b0[i]; bv[i + 8] = b1[i];
        }
        // D = A*B + C : emits v_wmma_f32_16x16x32_f16
        acc = __builtin_amdgcn_wmma_f32_16x16x32_f16(
                  false, av, false, bv, (short)0, acc, false, false);
    }

    // C/D layout: n = lane%16 ; VGPR r -> row m = r + 8*(lane/16)
    int nc    = (nt << 4) + (lane & 15);
    int mbase = (mt << 4) + ((lane >> 4) << 3);
#pragma unroll
    for (int r = 0; r < 8; ++r)
        C[(mbase + r) * G4H_ + nc] = acc[r];
}

// ---------------------------------------------------------------------------
// Kernel 3: Gx[b,g] = x[b,:] . W_ih0[g, 128:144] + b_ih0[g] + b_hh0[g]
// ---------------------------------------------------------------------------
__global__ void gx_kernel(const float* __restrict__ x,
                          const float* __restrict__ Wih0,
                          const float* __restrict__ bih0,
                          const float* __restrict__ bhh0,
                          float* __restrict__ Gx) {
    int idx = blockIdx.x * blockDim.x + threadIdx.x;   // 0..65535
    int b = idx >> 10, g = idx & 1023;
    float s = bih0[g] + bhh0[g];
    const float* xr = x + b * INCH;
    const float* wr = Wih0 + g * LIN0_ + FDIM_;
#pragma unroll
    for (int k = 0; k < INCH; ++k) s += xr[k] * wr[k];
    Gx[b * G4H_ + g] = s;
}

// ---------------------------------------------------------------------------
// Kernel 4: fused 3-layer projected-LSTM recurrence (P = 1 => h is scalar).
// One block per batch element; thread j owns c[j] and gate lanes
// {j, H+j, 2H+j, 3H+j}. h broadcast via shfl_xor + LDS reduction.
// ---------------------------------------------------------------------------
__device__ __forceinline__ float sigf(float x) {
    return 1.0f / (1.0f + __expf(-x));
}
__device__ __forceinline__ float tanhfast(float x) {
    return 2.0f / (1.0f + __expf(-2.0f * x)) - 1.0f;
}

__device__ __forceinline__ float block_sum(float v, float* red) {
#pragma unroll
    for (int off = 16; off > 0; off >>= 1)
        v += __shfl_xor(v, off, 32);
    int w = threadIdx.x >> 5;
    if ((threadIdx.x & 31) == 0) red[w] = v;
    __syncthreads();
    float s = red[0] + red[1] + red[2] + red[3] +
              red[4] + red[5] + red[6] + red[7];
    __syncthreads();
    return s;
}

__global__ void __launch_bounds__(256)
lstm_rec_kernel(const float* __restrict__ Gf,   const float* __restrict__ Gx,
                const float* __restrict__ Whh0, const float* __restrict__ Whr0,
                const float* __restrict__ Wih1, const float* __restrict__ Whh1,
                const float* __restrict__ bih1, const float* __restrict__ bhh1,
                const float* __restrict__ Whr1,
                const float* __restrict__ Wih2, const float* __restrict__ Whh2,
                const float* __restrict__ bih2, const float* __restrict__ bhh2,
                const float* __restrict__ Whr2,
                float* __restrict__ out) {
    __shared__ float red[8];
    const int b = blockIdx.x;
    const int j = threadIdx.x;

    // Layer 0 per-thread constants (resident in VGPRs across all 1001 steps)
    const float gx_i = Gx[b * G4H_ + j];
    const float gx_f = Gx[b * G4H_ + H_ + j];
    const float gx_g = Gx[b * G4H_ + 2 * H_ + j];
    const float gx_o = Gx[b * G4H_ + 3 * H_ + j];
    const float w0i = Whh0[j], w0f = Whh0[H_ + j];
    const float w0g = Whh0[2 * H_ + j], w0o = Whh0[3 * H_ + j];
    const float wr0 = Whr0[j];
    // Layer 1
    const float u1i = Wih1[j], u1f = Wih1[H_ + j];
    const float u1g = Wih1[2 * H_ + j], u1o = Wih1[3 * H_ + j];
    const float s1i = bih1[j] + bhh1[j],           s1f = bih1[H_ + j] + bhh1[H_ + j];
    const float s1g = bih1[2 * H_ + j] + bhh1[2 * H_ + j];
    const float s1o = bih1[3 * H_ + j] + bhh1[3 * H_ + j];
    const float w1i = Whh1[j], w1f = Whh1[H_ + j];
    const float w1g = Whh1[2 * H_ + j], w1o = Whh1[3 * H_ + j];
    const float wr1 = Whr1[j];
    // Layer 2
    const float u2i = Wih2[j], u2f = Wih2[H_ + j];
    const float u2g = Wih2[2 * H_ + j], u2o = Wih2[3 * H_ + j];
    const float s2i = bih2[j] + bhh2[j],           s2f = bih2[H_ + j] + bhh2[H_ + j];
    const float s2g = bih2[2 * H_ + j] + bhh2[2 * H_ + j];
    const float s2o = bih2[3 * H_ + j] + bhh2[3 * H_ + j];
    const float w2i = Whh2[j], w2f = Whh2[H_ + j];
    const float w2g = Whh2[2 * H_ + j], w2o = Whh2[3 * H_ + j];
    const float wr2 = Whr2[j];

    float c0 = 0.f, c1 = 0.f, c2 = 0.f;
    float h0 = 0.f, h1 = 0.f, h2 = 0.f;

    for (int t = 0; t < NF_; ++t) {
        const float* row = Gf + (size_t)t * G4H_;
        // hint next Gf row toward the WGP (global_prefetch_b8)
        if (t + 1 < NF_) __builtin_prefetch(row + G4H_ + j, 0, 0);

        // ---- layer 0 ----
        float gi = row[j]            + gx_i + h0 * w0i;
        float gf = row[H_ + j]       + gx_f + h0 * w0f;
        float gg = row[2 * H_ + j]   + gx_g + h0 * w0g;
        float go = row[3 * H_ + j]   + gx_o + h0 * w0o;
        c0 = sigf(gf) * c0 + sigf(gi) * tanhfast(gg);
        float hr = sigf(go) * tanhfast(c0);
        h0 = block_sum(hr * wr0, red);

        // ---- layer 1 (input dim P=1: scalar h0 * column) ----
        gi = h0 * u1i + s1i + h1 * w1i;
        gf = h0 * u1f + s1f + h1 * w1f;
        gg = h0 * u1g + s1g + h1 * w1g;
        go = h0 * u1o + s1o + h1 * w1o;
        c1 = sigf(gf) * c1 + sigf(gi) * tanhfast(gg);
        hr = sigf(go) * tanhfast(c1);
        h1 = block_sum(hr * wr1, red);

        // ---- layer 2 ----
        gi = h1 * u2i + s2i + h2 * w2i;
        gf = h1 * u2f + s2f + h2 * w2f;
        gg = h1 * u2g + s2g + h2 * w2g;
        go = h1 * u2o + s2o + h2 * w2o;
        c2 = sigf(gf) * c2 + sigf(gi) * tanhfast(gg);
        hr = sigf(go) * tanhfast(c2);
        h2 = block_sum(hr * wr2, red);

        if (j == 0) out[(size_t)b * NF_ + t] = h2;
    }
}

// ---------------------------------------------------------------------------
// Launch
// ---------------------------------------------------------------------------
extern "C" void kernel_launch(void* const* d_in, const int* in_sizes, int n_in,
                              void* d_out, int out_size, void* d_ws, size_t ws_size,
                              hipStream_t stream) {
    const float* x    = (const float*)d_in[0];
    const float* f    = (const float*)d_in[1];
    const float* Wih0 = (const float*)d_in[2];
    const float* Whh0 = (const float*)d_in[3];
    const float* bih0 = (const float*)d_in[4];
    const float* bhh0 = (const float*)d_in[5];
    const float* Whr0 = (const float*)d_in[6];
    const float* Wih1 = (const float*)d_in[7];
    const float* Whh1 = (const float*)d_in[8];
    const float* bih1 = (const float*)d_in[9];
    const float* bhh1 = (const float*)d_in[10];
    const float* Whr1 = (const float*)d_in[11];
    const float* Wih2 = (const float*)d_in[12];
    const float* Whh2 = (const float*)d_in[13];
    const float* bih2 = (const float*)d_in[14];
    const float* bhh2 = (const float*)d_in[15];
    const float* Whr2 = (const float*)d_in[16];

    // workspace layout (256B-aligned sections), ~4.75 MB total
    char* ws = (char*)d_ws;
    _Float16* f16f = (_Float16*)(ws);                       // 1024*128*2 = 256 KB
    _Float16* f16w = (_Float16*)(ws + 262144);              // 256 KB
    float*    Gf   = (float*)(ws + 524288);                 // 1024*1024*4 = 4 MB
    float*    Gx   = (float*)(ws + 524288 + 4194304);       // 64*1024*4 = 256 KB

    cvt_f16_kernel<<<1024, 256, 0, stream>>>(f, Wih0, f16f, f16w);
    wmma_gemm_kernel<<<512, 256, 0, stream>>>(f16f, f16w, Gf);
    gx_kernel<<<256, 256, 0, stream>>>(x, Wih0, bih0, bhh0, Gx);
    lstm_rec_kernel<<<B_, 256, 0, stream>>>(Gf, Gx, Whh0, Whr0,
                                            Wih1, Whh1, bih1, bhh1, Whr1,
                                            Wih2, Whh2, bih2, bhh2, Whr2,
                                            (float*)d_out);
}
